// GlowTTS_69861938037481
// MI455X (gfx1250) — matrix operand order repaired and checked by reading, not compiled, not run on hardware
//
#include <hip/hip_runtime.h>
#include <hip/hip_bf16.h>
#include <math.h>

// ---------------------------------------------------------------------------
// GlowTTS forward for MI455X (gfx1250, wave32, WMMA bf16 16x16x32)
// ---------------------------------------------------------------------------

typedef __bf16 v16bf __attribute__((ext_vector_type(16)));
typedef float  v8f   __attribute__((ext_vector_type(8)));

#define B_ 64
#define S_ 384
#define T_ 1536
#define C_ 80
#define D_ 192
#define V_ 256
#define NEGV (-1000000000.0f)

// ---- WMMA fragment builders (CDNA5 ISA 7.12.2 layouts) --------------------
// A (16x32 bf16): lane L -> row M = L&15; K set: L<16 ? {0..7,16..23} : {8..15,24..31}
static __device__ __forceinline__ v16bf
load_a_frag(const __bf16* lds, int stride, int row_base, int lane, int kchunk)
{
    int r  = row_base + (lane & 15);
    int kb = kchunk * 32 + ((lane & 16) ? 8 : 0);
    const __bf16* p = lds + r * stride + kb;
    v16bf a;
#pragma unroll
    for (int j = 0; j < 8; ++j) a[j] = p[j];
#pragma unroll
    for (int j = 0; j < 8; ++j) a[8 + j] = p[16 + j];
    return a;
}

// B (32x16 bf16): lane L -> col N = L&15; K set: L<16 ? {0..7,16..23} : {8..15,24..31}
static __device__ __forceinline__ v16bf
load_b_frag(const __bf16* lds, int stride, int lane, int kchunk, int ncol)
{
    const __bf16* p = lds + (kchunk * 32 + ((lane & 16) ? 8 : 0)) * stride + ncol + (lane & 15);
    v16bf b;
#pragma unroll
    for (int j = 0; j < 8; ++j) b[j] = p[j * stride];
#pragma unroll
    for (int j = 0; j < 8; ++j) b[8 + j] = p[(16 + j) * stride];
    return b;
}

static __device__ __forceinline__ v8f wmma_bf16(v16bf a, v16bf b, v8f c)
{
    return __builtin_amdgcn_wmma_f32_16x16x32_bf16(false, a, false, b,
                                                   (short)0, c, false, false);
}

// ---------------------------------------------------------------------------
// Kernel 1: src_latent = tanh(emb[src] @ W_enc)     M=B*S=24576, N=K=192
// 128 threads (4 waves); WG computes 64 rows x 192 cols. W staged per-K-chunk.
// ---------------------------------------------------------------------------
#define AST 200  // (D_+8) padded LDS stride for A tiles

__global__ __launch_bounds__(128) void k_encoder(const int* __restrict__ src,
                                                 const float* __restrict__ emb,
                                                 const float* __restrict__ Wenc,
                                                 float* __restrict__ out_latent)
{
    __shared__ __bf16 sA[64 * AST];
    __shared__ __bf16 sW[32 * AST];   // one K-chunk of W (32 x 192)
    __shared__ int sIdx[64];

    const int tid = threadIdx.x;
    const int m0  = blockIdx.x * 64;

    if (tid < 64) sIdx[tid] = src[m0 + tid];
    __syncthreads();
    for (int r = 0; r < 64; ++r) {
        int idx = sIdx[r];
        for (int k = tid; k < D_; k += 128)
            sA[r * AST + k] = (__bf16)emb[idx * D_ + k];
    }

    const int lane = tid & 31;
    const int rb   = (tid >> 5) * 16;
    const v8f vz = {0.f, 0.f, 0.f, 0.f, 0.f, 0.f, 0.f, 0.f};
    v8f acc[12];
#pragma unroll
    for (int n = 0; n < 12; ++n) acc[n] = vz;

    for (int kt = 0; kt < 6; ++kt) {
        __syncthreads();   // protect previous chunk use + A staging (first iter)
        for (int i = tid; i < 32 * D_; i += 128) {
            int kk = i / D_, n = i % D_;
            sW[kk * AST + n] = (__bf16)Wenc[(kt * 32 + kk) * D_ + n];
        }
        __syncthreads();
        v16bf a = load_a_frag(sA, AST, rb, lane, kt);
#pragma unroll
        for (int nt = 0; nt < 12; ++nt) {
            v16bf b = load_b_frag(sW, AST, lane, 0, nt * 16);
            acc[nt] = wmma_bf16(a, b, acc[nt]);
        }
    }

    const int mrow = m0 + rb + ((lane & 16) ? 8 : 0);
    const int ncol = lane & 15;
#pragma unroll
    for (int nt = 0; nt < 12; ++nt)
#pragma unroll
        for (int r8 = 0; r8 < 8; ++r8)
            out_latent[(size_t)(mrow + r8) * D_ + nt * 16 + ncol] =
                tanhf(acc[nt][r8]);
}

// ---------------------------------------------------------------------------
// Kernel 2: projections. N=160 ([W_mean | W_logstd]), K=192.
// Produces Acoef=-0.5*inv_var, Bm=mean*inv_var, cterm, log_est_duration.
// ---------------------------------------------------------------------------
#define WST2 168  // 160+8 padded stride

__global__ __launch_bounds__(128) void k_project(const float* __restrict__ latent,
                                                 const float* __restrict__ Wmean,
                                                 const float* __restrict__ Wls,
                                                 const float* __restrict__ wdur,
                                                 const int* __restrict__ src_len,
                                                 float* __restrict__ wAc,
                                                 float* __restrict__ wBm,
                                                 float* __restrict__ wCt,
                                                 float* __restrict__ out_led)
{
    __shared__ __bf16 sA[64 * AST];
    __shared__ __bf16 sW[32 * WST2];
    __shared__ float sDur[2][64];
    __shared__ float sCt[64];

    const int tid = threadIdx.x;
    const int m0  = blockIdx.x * 64;

    for (int i = tid; i < 64 * D_; i += 128) {
        int r = i / D_, k = i % D_;
        sA[r * AST + k] = (__bf16)latent[(size_t)(m0 + r) * D_ + k];
    }
    {   // deterministic w_dur dot: 2 threads per row, fixed halves
        int r = tid & 63, h = tid >> 6;
        float p = 0.f;
        for (int k = 0; k < 96; ++k)
            p += latent[(size_t)(m0 + r) * D_ + h * 96 + k] * wdur[h * 96 + k];
        sDur[h][r] = p;
    }

    const int lane = tid & 31;
    const int rb   = (tid >> 5) * 16;
    const v8f vz = {0.f, 0.f, 0.f, 0.f, 0.f, 0.f, 0.f, 0.f};
    v8f acc[10];
#pragma unroll
    for (int n = 0; n < 10; ++n) acc[n] = vz;

    for (int kt = 0; kt < 6; ++kt) {
        __syncthreads();
        for (int i = tid; i < 32 * C_; i += 128) {
            int kk = i / C_, n = i % C_;
            int kg = kt * 32 + kk;
            sW[kk * WST2 + n]      = (__bf16)Wmean[kg * C_ + n];
            sW[kk * WST2 + 80 + n] = (__bf16)Wls[kg * C_ + n];
        }
        __syncthreads();
        v16bf a = load_a_frag(sA, AST, rb, lane, kt);
#pragma unroll
        for (int nt = 0; nt < 10; ++nt) {
            v16bf b = load_b_frag(sW, WST2, lane, 0, nt * 16);
            acc[nt] = wmma_bf16(a, b, acc[nt]);
        }
    }

    const int half8 = (lane & 16) ? 8 : 0;
    const int ncol  = lane & 15;
    const float LOG2PI_TERM = 73.51508265637381f;  // 0.5*C*ln(2*pi)
#pragma unroll
    for (int r8 = 0; r8 < 8; ++r8) {
        int mloc = rb + r8 + half8;
        size_t mg = (size_t)(m0 + mloc);
        float ct = 0.f;
#pragma unroll
        for (int nt = 0; nt < 5; ++nt) {
            float mean = acc[nt][r8];
            float ls   = tanhf(acc[nt + 5][r8]);
            float iv   = expf(-2.f * ls);
            int c = nt * 16 + ncol;
            wAc[mg * C_ + c] = -0.5f * iv;
            wBm[mg * C_ + c] = mean * iv;
            ct += -0.5f * mean * mean * iv - ls;
        }
        // reduce across the 16 lanes of this half (xor masks stay in-half)
        ct += __shfl_xor(ct, 1, 32);
        ct += __shfl_xor(ct, 2, 32);
        ct += __shfl_xor(ct, 4, 32);
        ct += __shfl_xor(ct, 8, 32);
        if (ncol == 0) sCt[mloc] = ct - LOG2PI_TERM;
    }
    __syncthreads();
    if (tid < 64) {
        int m = m0 + tid;
        int b = m / S_, s = m % S_;
        wCt[m] = sCt[tid];
        float dur = sDur[0][tid] + sDur[1][tid];
        out_led[m] = (s >= src_len[b]) ? -INFINITY : dur;
    }
}

// ---------------------------------------------------------------------------
// Kernel 3: z = where(t>=tgt_len, 0, transpose(tgt)*exp(log_a)+b_aff)
// ---------------------------------------------------------------------------
__global__ __launch_bounds__(256) void k_z(const float* __restrict__ tgt,
                                           const int* __restrict__ tgt_len,
                                           const float* __restrict__ log_a,
                                           const float* __restrict__ b_aff,
                                           float* __restrict__ out_z)
{
    __shared__ float tile[C_][65];
    __shared__ float sLa[C_], sBa[C_];
    const int tid = threadIdx.x;
    const int t0  = blockIdx.x * 64;
    const int b   = blockIdx.y;
    const int tl  = tgt_len[b];
    if (tid < C_) { sLa[tid] = expf(log_a[tid]); sBa[tid] = b_aff[tid]; }
    for (int i = tid; i < C_ * 64; i += 256) {
        int c = i >> 6, t = i & 63;
        tile[c][t] = tgt[((size_t)b * C_ + c) * T_ + t0 + t];
    }
    __syncthreads();
    for (int i = tid; i < 64 * C_; i += 256) {
        int t = i / C_, c = i % C_;
        int tg = t0 + t;
        float v = (tg >= tl) ? 0.f : tile[c][t] * sLa[c] + sBa[c];
        out_z[((size_t)b * T_ + tg) * C_ + c] = v;
    }
}

// ---------------------------------------------------------------------------
// Kernel 4: log_prob[b,t,s] = sum_c z2*Acoef + z*Bm + cterm, masked to NEG.
// Batched WMMA GEMM, K padded 80->96. WG = 64t x 64s, 4 waves.
// ---------------------------------------------------------------------------
#define ZST 104  // 96+8
#define BST 72   // 64+8

__global__ __launch_bounds__(128) void k_logprob(const float* __restrict__ zg,
                                                 const float* __restrict__ wAc,
                                                 const float* __restrict__ wBm,
                                                 const float* __restrict__ wCt,
                                                 const int* __restrict__ src_len,
                                                 const int* __restrict__ tgt_len,
                                                 float* __restrict__ lp)
{
    __shared__ __bf16 sZ[64 * ZST], sZ2[64 * ZST];
    __shared__ __bf16 sAc[96 * BST], sBm[96 * BST];
    __shared__ float sCt[64];

    const int tid = threadIdx.x;
    const int s0 = blockIdx.x * 64, t0 = blockIdx.y * 64, b = blockIdx.z;
    const int sl = src_len[b], tl = tgt_len[b];

    for (int i = tid; i < 64 * 96; i += 128) {
        int t = i / 96, c = i % 96;
        float v = (c < C_) ? zg[((size_t)b * T_ + t0 + t) * C_ + c] : 0.f;
        sZ [t * ZST + c] = (__bf16)v;
        sZ2[t * ZST + c] = (__bf16)(v * v);
    }
    for (int i = tid; i < 64 * 96; i += 128) {
        int s = i / 96, c = i % 96;
        float a = 0.f, m = 0.f;
        if (c < C_) {
            size_t off = ((size_t)b * S_ + s0 + s) * C_ + c;
            a = wAc[off]; m = wBm[off];
        }
        sAc[c * BST + s] = (__bf16)a;   // transposed: LDS is [c][s]
        sBm[c * BST + s] = (__bf16)m;
    }
    if (tid < 64) sCt[tid] = wCt[b * S_ + s0 + tid];
    __syncthreads();

    const int lane = tid & 31;
    const int rb   = (tid >> 5) * 16;
    const v8f vz = {0.f, 0.f, 0.f, 0.f, 0.f, 0.f, 0.f, 0.f};
    v8f acc[4] = {vz, vz, vz, vz};

#pragma unroll
    for (int kt = 0; kt < 3; ++kt) {            // z^2 * (-0.5*inv_var)
        v16bf a = load_a_frag(sZ2, ZST, rb, lane, kt);
#pragma unroll
        for (int nt = 0; nt < 4; ++nt) {
            v16bf bb = load_b_frag(sAc, BST, lane, kt, nt * 16);
            acc[nt] = wmma_bf16(a, bb, acc[nt]);
        }
    }
#pragma unroll
    for (int kt = 0; kt < 3; ++kt) {            // z * (mean*inv_var)
        v16bf a = load_a_frag(sZ, ZST, rb, lane, kt);
#pragma unroll
        for (int nt = 0; nt < 4; ++nt) {
            v16bf bb = load_b_frag(sBm, BST, lane, kt, nt * 16);
            acc[nt] = wmma_bf16(a, bb, acc[nt]);
        }
    }

    const int trow = t0 + rb + ((lane & 16) ? 8 : 0);
#pragma unroll
    for (int nt = 0; nt < 4; ++nt) {
        int sg = s0 + nt * 16 + (lane & 15);
        float ct = sCt[nt * 16 + (lane & 15)];
        bool smask = (sg >= sl);
#pragma unroll
        for (int r8 = 0; r8 < 8; ++r8) {
            int tg = trow + r8;
            float v = acc[nt][r8] + ct;
            if (smask || tg >= tl) v = NEGV;
            lp[((size_t)b * T_ + tg) * S_ + sg] = v;
        }
    }
}

// ---------------------------------------------------------------------------
// Kernel 5: MAS. One workgroup per batch, 384 threads (=S, 12 waves).
// DP with wave32 ballot for choice bits, then backtrack + durations + logdet.
// ---------------------------------------------------------------------------
__global__ __launch_bounds__(384) void k_mas(const float* __restrict__ lp,
                                             unsigned* __restrict__ wCh,
                                             const int* __restrict__ src_len,
                                             const int* __restrict__ tgt_len,
                                             const float* __restrict__ log_a,
                                             float* __restrict__ out_dur,
                                             float* __restrict__ out_logdet)
{
    __shared__ float dpA[S_], dpB[S_];
    __shared__ unsigned short path[T_];
    __shared__ int dur[S_];
    __shared__ float red[S_];
    __shared__ float sSla;

    const int b = blockIdx.x, tid = threadIdx.x;
    const int sl = src_len[b], tl = tgt_len[b];
    const float* lpb = lp + (size_t)b * T_ * S_;
    unsigned* chb = wCh + (size_t)b * T_ * 12;

    dpA[tid] = (tid == 0) ? 0.f : NEGV;
    dur[tid] = 0;
    if (tid == 0) {
        float s = 0.f;
        for (int c = 0; c < C_; ++c) s += log_a[c];
        sSla = s;
    }
    __syncthreads();

    float* cur = dpA;
    float* nxt = dpB;
    const int wv = tid >> 5;
    for (int t = 0; t < T_; ++t) {
        float l = lpb[(size_t)t * S_ + tid];
        if (t + 8 < T_)
            __builtin_prefetch(&lpb[(size_t)(t + 8) * S_ + tid], 0, 3);
        float c0 = cur[tid];
        float sh = (tid == 0) ? NEGV : cur[tid - 1];
        bool take = sh > c0;
        unsigned m = __builtin_amdgcn_ballot_w32(take);
        if ((tid & 31) == 0) chb[t * 12 + wv] = m;
        nxt[tid] = l + (take ? sh : c0);
        __syncthreads();
        float* tmp = cur; cur = nxt; nxt = tmp;
    }

    if (tid == 0) {                       // backtrack (choices L2-resident)
        int scur = sl - 1;
        for (int t = T_ - 1; t >= 0; --t) {
            if (t == tl - 1) scur = sl - 1;
            path[t] = (unsigned short)scur;
            if (t < tl) {
                unsigned mm = chb[t * 12 + (scur >> 5)];
                scur -= (int)((mm >> (scur & 31)) & 1u);
                if (scur < 0) scur = 0;
            }
        }
    }
    __syncthreads();

    float part = 0.f;
    for (int t = tid; t < T_; t += S_) {   // 4 iterations
        if (t < tl) {
            int s = path[t];
            atomicAdd(&dur[s], 1);         // integer: order-deterministic
            part += lpb[(size_t)t * S_ + s];
        }
    }
    red[tid] = part;
    __syncthreads();
    if (tid < 128) red[tid] += red[tid + 128] + red[tid + 256];
    __syncthreads();
    for (int off = 64; off > 0; off >>= 1) {
        if (tid < off) red[tid] += red[tid + off];
        __syncthreads();
    }
    out_dur[b * S_ + tid] = (float)dur[tid];
    if (tid == 0) out_logdet[b] = red[0] + (float)tl * sSla;
}

// ---------------------------------------------------------------------------
extern "C" void kernel_launch(void* const* d_in, const int* in_sizes, int n_in,
                              void* d_out, int out_size, void* d_ws, size_t ws_size,
                              hipStream_t stream)
{
    const int*   src   = (const int*)d_in[0];
    const float* tgt   = (const float*)d_in[1];
    const int*   slen  = (const int*)d_in[2];
    const int*   tlen  = (const int*)d_in[3];
    const float* emb   = (const float*)d_in[4];
    const float* Wenc  = (const float*)d_in[5];
    const float* Wmean = (const float*)d_in[6];
    const float* Wls   = (const float*)d_in[7];
    const float* wdur  = (const float*)d_in[8];
    const float* loga  = (const float*)d_in[9];
    const float* baff  = (const float*)d_in[10];

    // Output layout (tuple flattened in return order, f32):
    float* out      = (float*)d_out;
    float* o_latent = out;                               // B*S*D
    float* o_z      = o_latent + (size_t)B_ * S_ * D_;   // B*T*C
    float* o_led    = o_z + (size_t)B_ * T_ * C_;        // B*S
    float* o_dur    = o_led + (size_t)B_ * S_;           // B*S
    float* o_logdet = o_dur + (size_t)B_ * S_;           // B

    // Workspace layout (~172 MB):
    float* ws  = (float*)d_ws;
    float* wAc = ws;                                     // B*S*C
    float* wBm = wAc + (size_t)B_ * S_ * C_;             // B*S*C
    float* wCt = wBm + (size_t)B_ * S_ * C_;             // B*S
    float* wLp = wCt + (size_t)B_ * S_;                  // B*T*S
    unsigned* wCh = (unsigned*)(wLp + (size_t)B_ * T_ * S_);  // B*T*12

    k_encoder<<<(B_ * S_) / 64, 128, 0, stream>>>(src, emb, Wenc, o_latent);
    k_project<<<(B_ * S_) / 64, 128, 0, stream>>>(o_latent, Wmean, Wls, wdur,
                                                  slen, wAc, wBm, wCt, o_led);
    k_z<<<dim3(T_ / 64, B_), 256, 0, stream>>>(tgt, tlen, loga, baff, o_z);
    k_logprob<<<dim3(S_ / 64, T_ / 64, B_), 128, 0, stream>>>(o_z, wAc, wBm, wCt,
                                                              slen, tlen, wLp);
    k_mas<<<B_, S_, 0, stream>>>(wLp, wCh, slen, tlen, loga, o_dur, o_logdet);
}
